// AttentionDvit_57320633533121
// MI455X (gfx1250) — matrix operand order repaired and checked by reading, not compiled, not run on hardware
//
#include <hip/hip_runtime.h>

// ---------------------------------------------------------------------------
// DynamicViT attention forward for gfx1250 (MI455X), wave32 + WMMA f16.
// Memory-bound on the 256MB softmax output; all GEMMs via v_wmma_f32_16x16x32_f16.
// The 256MB probs output is written with NT (non-temporal) stores so it does
// not evict the reused K/V/Q working set from the 192MB L2.
// Workspace layout (bytes, all 256-aligned):
//   xh     [B*N, DIM]      f16   @ 0          (8 MB)
//   wqkvt  [1536, 512]     f16   @ 8388608    (1.5 MB)
//   woutt  [512, 512]      f16   @ 9961472    (0.5 MB)
//   Qh     [B,H,N,DH]      f16   @ 10485760   (8 MB)  (pre-scaled by DH^-0.5)
//   Kh     [B,H,N,DH]      f16   @ 18874368   (8 MB)
//   Vt     [B,H,DH,N]      f16   @ 27262976   (8 MB)  (V transposed)
//   Oh     [B,N,H*DH]      f16   @ 35651584   (8 MB)
// Total ~42 MB.
// ---------------------------------------------------------------------------

typedef __attribute__((ext_vector_type(16))) _Float16 v16h;
typedef __attribute__((ext_vector_type(8)))  _Float16 v8h;
typedef __attribute__((ext_vector_type(8)))  float    v8f;
typedef __attribute__((ext_vector_type(2)))  float    v2f;

constexpr int Bq   = 8;
constexpr int Nq   = 1024;
constexpr int DIMq = 512;
constexpr int Hq   = 8;
constexpr int DHq  = 64;
constexpr int QKVC = 3 * Hq * DHq;   // 1536
constexpr float SCALEq = 0.125f;     // DH^-0.5, exact power of two
constexpr float EPSq   = 1e-6f;
constexpr float EPSNq  = EPSq / (float)Nq;

#define DEV __device__ __forceinline__

// A fragment: 16x32 (MxK) f16, row-major source with row stride ld (halves).
// ISA layout: lanes 0-15/16-31 both M=lane&15; halves [0..7]=K 8*hi+j,
// halves [8..15]=K 16+8*hi+j.
DEV v16h load_A16(const _Float16* __restrict__ src, int ld, int lane) {
  const int lo = lane & 15, hi = lane >> 4;
  const v8h x0 = *(const v8h*)(src + lo * ld + 8 * hi);
  const v8h x1 = *(const v8h*)(src + lo * ld + 16 + 8 * hi);
  v16h a;
#pragma unroll
  for (int j = 0; j < 8; ++j) { a[j] = x0[j]; a[8 + j] = x1[j]; }
  return a;
}

// B fragment: 32x16 (KxN) f16, read from a TRANSPOSED source laid out as
// X[n][k] row-major with row stride ld, so k' runs contiguous per lane.
// ISA layout: N = lane&15; halves [0..7] = K 8*hi + j, halves [8..15] = K 16+8*hi+j.
DEV v16h load_Bt16(const _Float16* __restrict__ src, int ld, int lane) {
  const int n = lane & 15, hi = lane >> 4;
  const v8h x0 = *(const v8h*)(src + n * ld + 8 * hi);
  const v8h x1 = *(const v8h*)(src + n * ld + 16 + 8 * hi);
  v16h b;
#pragma unroll
  for (int j = 0; j < 8; ++j) { b[j] = x0[j]; b[8 + j] = x1[j]; }
  return b;
}

DEV v8f wmma16(v16h a, v16h b, v8f c) {
  return __builtin_amdgcn_wmma_f32_16x16x32_f16(
      false, a, false, b, (short)0, c, false, false);
}

DEV v8f zero8() { v8f z = {0.f, 0.f, 0.f, 0.f, 0.f, 0.f, 0.f, 0.f}; return z; }

// ---------------------------------------------------------------------------
// Conversion / transpose kernels (one-shot streaming: NT loads)
// ---------------------------------------------------------------------------
__global__ __launch_bounds__(256) void k_cvt_x(const float* __restrict__ x,
                                               _Float16* __restrict__ xh) {
  int i = blockIdx.x * 256 + threadIdx.x;   // grid sized exactly to B*N*DIM
  xh[i] = (_Float16)__builtin_nontemporal_load(x + i);
}

// Wt[o*rows + d] = W[d*cols + o]   (rows = DIM, cols = out-features)
__global__ __launch_bounds__(256) void k_transpose_w(const float* __restrict__ W,
                                                     _Float16* __restrict__ Wt,
                                                     int rows, int cols) {
  int i = blockIdx.x * 256 + threadIdx.x;
  int o = i / rows, d = i - o * rows;
  Wt[i] = (_Float16)W[(size_t)d * cols + o];
}

// ---------------------------------------------------------------------------
// QKV projection: [B*N,512] x [512,1536] -> Q/K scaled f16, V transposed f16
// 256 threads = 8 waves; 64x64 tile per WG; 2 accumulators per wave.
// ---------------------------------------------------------------------------
__global__ __launch_bounds__(256) void k_gemm_qkv(
    const _Float16* __restrict__ Xh, const _Float16* __restrict__ Wt,
    _Float16* __restrict__ Qh, _Float16* __restrict__ Kh,
    _Float16* __restrict__ Vt) {
  const int m0 = blockIdx.x * 64, o0 = blockIdx.y * 64;
  const int tid = threadIdx.x, wave = tid >> 5, lane = tid & 31;
  const int lo = lane & 15, hi = lane >> 4;
  const int mi = wave >> 1, ni0 = (wave & 1) * 2;

  v8f acc0 = zero8(), acc1 = zero8();
  const _Float16* Arow = Xh + (m0 + mi * 16) * DIMq;
  const _Float16* B0   = Wt + (o0 + ni0 * 16) * DIMq;
  const _Float16* B1   = Wt + (o0 + ni0 * 16 + 16) * DIMq;
#pragma unroll 4
  for (int k0 = 0; k0 < DIMq; k0 += 32) {
    v16h a  = load_A16(Arow + k0, DIMq, lane);
    v16h b0 = load_Bt16(B0 + k0, DIMq, lane);
    v16h b1 = load_Bt16(B1 + k0, DIMq, lane);
    acc0 = wmma16(a, b0, acc0);
    acc1 = wmma16(a, b1, acc1);
  }

#pragma unroll
  for (int t = 0; t < 2; ++t) {
    const v8f& acc = t ? acc1 : acc0;
    const int o  = o0 + (ni0 + t) * 16 + lo;
    const int sec = o >> 9, oc = o & 511, hh = oc >> 6, dd = oc & 63;
#pragma unroll
    for (int r = 0; r < 8; ++r) {
      const int m = m0 + mi * 16 + 8 * hi + r;
      const int bb = m >> 10, nn = m & 1023, bh = bb * Hq + hh;
      const float v = acc[r];
      if (sec == 0)       Qh[(size_t)(bh * Nq + nn) * DHq + dd] = (_Float16)(v * SCALEq);
      else if (sec == 1)  Kh[(size_t)(bh * Nq + nn) * DHq + dd] = (_Float16)v;
      else                Vt[(size_t)(bh * DHq + dd) * Nq + nn] = (_Float16)v;
    }
  }
}

// ---------------------------------------------------------------------------
// Attention: one WG = 16 query rows of one (b,h); 128 threads = 4 waves.
// Pass1: QK^T via WMMA -> row max (S recomputed instead of stored: LDS-cheap).
// Pass2: QK^T again -> e=exp(s-max) f16 in LDS, plain & policy-masked sums.
// Writes plain softmax (the 256MB output) with NT stores, then P@V via WMMA
// with P fragments synthesized on the fly from the LDS exp tile.
// ---------------------------------------------------------------------------
__global__ __launch_bounds__(128) void k_attn(
    const _Float16* __restrict__ Qh, const _Float16* __restrict__ Kh,
    const _Float16* __restrict__ Vt, const float* __restrict__ policy,
    float* __restrict__ out, _Float16* __restrict__ Oh) {
  constexpr int EPAD = 1032;                 // pad to dodge LDS bank conflicts
  __shared__ _Float16 Esm[16 * EPAD];        // exp(s - rowmax), f16
  __shared__ float pol[Nq];
  __shared__ float wred[3][4][16];           // per-wave partial max/sum/sumpol
  __shared__ float rstat[3][16];             // rowmax, 1/sum_plain, 1/(sum_pol+eps)

  const int bh = blockIdx.y, b = bh >> 3, h = bh & 7;
  const int m0 = blockIdx.x * 16;
  const int tid = threadIdx.x, wave = tid >> 5, lane = tid & 31;
  const int lo = lane & 15, hi = lane >> 4;

  for (int c = tid; c < Nq; c += 128) pol[c] = policy[b * Nq + c];

  const _Float16* Qbase = Qh + (size_t)(bh * Nq + m0) * DHq;  // pre-scaled Q
  const v16h aq0 = load_A16(Qbase, DHq, lane);
  const v16h aq1 = load_A16(Qbase + 32, DHq, lane);
  __syncthreads();

  // ---- pass 1: row maxima over raw scaled dots -------------------------
  float mx[8];
#pragma unroll
  for (int r = 0; r < 8; ++r) mx[r] = -1e30f;
  for (int nt = wave * 16; nt < wave * 16 + 16; ++nt) {
    const _Float16* Kb = Kh + (size_t)(bh * Nq + nt * 16) * DHq;
    v16h b0 = load_Bt16(Kb, DHq, lane);
    v16h b1 = load_Bt16(Kb + 32, DHq, lane);
    v8f s = zero8();
    s = wmma16(aq0, b0, s);
    s = wmma16(aq1, b1, s);
#pragma unroll
    for (int r = 0; r < 8; ++r) mx[r] = fmaxf(mx[r], s[r]);
  }
#pragma unroll
  for (int r = 0; r < 8; ++r)
#pragma unroll
    for (int m = 1; m < 16; m <<= 1)
      mx[r] = fmaxf(mx[r], __shfl_xor(mx[r], m, 32));
  if (lo == 0)
#pragma unroll
    for (int r = 0; r < 8; ++r) wred[0][wave][8 * hi + r] = mx[r];
  __syncthreads();
  if (tid < 16)
    rstat[0][tid] = fmaxf(fmaxf(wred[0][0][tid], wred[0][1][tid]),
                          fmaxf(wred[0][2][tid], wred[0][3][tid]));
  __syncthreads();

  // ---- pass 2: recompute S, exponentiate, accumulate sums --------------
  float rm[8], sp[8], sq[8];
#pragma unroll
  for (int r = 0; r < 8; ++r) { rm[r] = rstat[0][8 * hi + r]; sp[r] = 0.f; sq[r] = 0.f; }
  for (int nt = wave * 16; nt < wave * 16 + 16; ++nt) {
    const int n0 = nt * 16;
    const _Float16* Kb = Kh + (size_t)(bh * Nq + n0) * DHq;
    v16h b0 = load_Bt16(Kb, DHq, lane);
    v16h b1 = load_Bt16(Kb + 32, DHq, lane);
    v8f s = zero8();
    s = wmma16(aq0, b0, s);
    s = wmma16(aq1, b1, s);
    const int col = n0 + lo;
    const float keepc = pol[col];
#pragma unroll
    for (int r = 0; r < 8; ++r) {
      const int row = 8 * hi + r;
      const float e = __expf(s[r] - rm[r]);
      Esm[row * EPAD + col] = (_Float16)e;
      sp[r] += e;
      sq[r] += e * ((col == m0 + row) ? 1.f : keepc);  // always attend to self
    }
  }
#pragma unroll
  for (int r = 0; r < 8; ++r)
#pragma unroll
    for (int m = 1; m < 16; m <<= 1) {
      sp[r] += __shfl_xor(sp[r], m, 32);
      sq[r] += __shfl_xor(sq[r], m, 32);
    }
  if (lo == 0)
#pragma unroll
    for (int r = 0; r < 8; ++r) {
      wred[1][wave][8 * hi + r] = sp[r];
      wred[2][wave][8 * hi + r] = sq[r];
    }
  __syncthreads();
  if (tid < 16) {
    float s1 = wred[1][0][tid] + wred[1][1][tid] + wred[1][2][tid] + wred[1][3][tid];
    float s2 = wred[2][0][tid] + wred[2][1][tid] + wred[2][2][tid] + wred[2][3][tid];
    rstat[1][tid] = 1.f / s1;
    rstat[2][tid] = 1.f / (s2 + EPSq);
  }
  __syncthreads();

  // ---- plain softmax output: 256MB stream, written once, never re-read.
  // float2 per lane + NT store (TH=NT) so it bypasses L2 retention and the
  // K/V/Q/E working set stays resident in the 192MB L2.
  float* out1 = out + (size_t)Bq * Nq * DIMq + (size_t)(bh * Nq + m0) * Nq;
  for (int r = 0; r < 16; ++r) {
    const float inv = rstat[1][r];
    for (int c = tid * 2; c < Nq; c += 256) {
      v2f pv;
      pv[0] = (float)Esm[r * EPAD + c]     * inv;
      pv[1] = (float)Esm[r * EPAD + c + 1] * inv;
      __builtin_nontemporal_store(pv, (v2f*)(out1 + (size_t)r * Nq + c));
    }
  }

  // ---- P @ V : each wave owns one 16x16 output tile (DH=64 = 4 tiles) --
  const float invq = rstat[2][lo];
  const float addt = EPSNq * invq;
  const int n0 = wave * 16;
  const _Float16* Vb = Vt + (size_t)(bh * DHq + n0) * Nq;
  const int diagk = m0 + lo;
  v8f acc = zero8();
  for (int k0 = 0; k0 < Nq; k0 += 32) {
    const v8h ea = *(const v8h*)(Esm + lo * EPAD + k0 + 8 * hi);
    const v8h eb = *(const v8h*)(Esm + lo * EPAD + k0 + 16 + 8 * hi);
    v16h a;
#pragma unroll
    for (int j = 0; j < 8; ++j) {
      const int k1 = k0 + 8 * hi + j;
      const float kp1 = (k1 == diagk) ? 1.f : pol[k1];
      a[j] = (_Float16)((float)ea[j] * kp1 * invq + addt);
      const int k2 = k1 + 16;
      const float kp2 = (k2 == diagk) ? 1.f : pol[k2];
      a[8 + j] = (_Float16)((float)eb[j] * kp2 * invq + addt);
    }
    v16h bf = load_Bt16(Vb + k0, Nq, lane);
    acc = wmma16(a, bf, acc);
  }
#pragma unroll
  for (int r = 0; r < 8; ++r) {
    const int row = 8 * hi + r;
    Oh[(size_t)(b * Nq + m0 + row) * (Hq * DHq) + h * DHq + n0 + lo] = (_Float16)acc[r];
  }
}

// ---------------------------------------------------------------------------
// Output projection: [B*N,512] x [512,512] + bias -> f32 out
// ---------------------------------------------------------------------------
__global__ __launch_bounds__(256) void k_gemm_out(
    const _Float16* __restrict__ Oh, const _Float16* __restrict__ Wt,
    const float* __restrict__ bias, float* __restrict__ out0) {
  const int m0 = blockIdx.x * 64, o0 = blockIdx.y * 64;
  const int tid = threadIdx.x, wave = tid >> 5, lane = tid & 31;
  const int lo = lane & 15, hi = lane >> 4;
  const int mi = wave >> 1, ni0 = (wave & 1) * 2;

  v8f acc0 = zero8(), acc1 = zero8();
  const _Float16* Arow = Oh + (size_t)(m0 + mi * 16) * DIMq;
  const _Float16* B0   = Wt + (o0 + ni0 * 16) * DIMq;
  const _Float16* B1   = Wt + (o0 + ni0 * 16 + 16) * DIMq;
#pragma unroll 4
  for (int k0 = 0; k0 < DIMq; k0 += 32) {
    v16h a  = load_A16(Arow + k0, DIMq, lane);
    v16h b0 = load_Bt16(B0 + k0, DIMq, lane);
    v16h b1 = load_Bt16(B1 + k0, DIMq, lane);
    acc0 = wmma16(a, b0, acc0);
    acc1 = wmma16(a, b1, acc1);
  }
#pragma unroll
  for (int t = 0; t < 2; ++t) {
    const v8f& acc = t ? acc1 : acc0;
    const int o = o0 + (ni0 + t) * 16 + lo;
    const float bo = bias[o];
#pragma unroll
    for (int r = 0; r < 8; ++r) {
      const int m = m0 + mi * 16 + 8 * hi + r;
      out0[(size_t)m * DIMq + o] = acc[r] + bo;
    }
  }
}

// ---------------------------------------------------------------------------
extern "C" void kernel_launch(void* const* d_in, const int* in_sizes, int n_in,
                              void* d_out, int out_size, void* d_ws, size_t ws_size,
                              hipStream_t stream) {
  (void)in_sizes; (void)n_in; (void)out_size; (void)ws_size;
  const float* x      = (const float*)d_in[0];
  const float* policy = (const float*)d_in[1];
  const float* W_qkv  = (const float*)d_in[2];
  const float* W_out  = (const float*)d_in[3];
  const float* b_out  = (const float*)d_in[4];
  float* out = (float*)d_out;

  uint8_t* ws = (uint8_t*)d_ws;
  _Float16* xh    = (_Float16*)(ws + 0);
  _Float16* wqkvt = (_Float16*)(ws + 8388608);
  _Float16* woutt = (_Float16*)(ws + 9961472);
  _Float16* Qh    = (_Float16*)(ws + 10485760);
  _Float16* Kh    = (_Float16*)(ws + 18874368);
  _Float16* Vt    = (_Float16*)(ws + 27262976);
  _Float16* Oh    = (_Float16*)(ws + 35651584);

  k_cvt_x<<<(Bq * Nq * DIMq) / 256, 256, 0, stream>>>(x, xh);
  k_transpose_w<<<(DIMq * QKVC) / 256, 256, 0, stream>>>(W_qkv, wqkvt, DIMq, QKVC);
  k_transpose_w<<<(DIMq * DIMq) / 256, 256, 0, stream>>>(W_out, woutt, DIMq, DIMq);

  k_gemm_qkv<<<dim3(Bq * Nq / 64, QKVC / 64), 256, 0, stream>>>(xh, wqkvt, Qh, Kh, Vt);

  k_attn<<<dim3(Nq / 16, Bq * Hq), 128, 0, stream>>>(Qh, Kh, Vt, policy, out, Oh);

  k_gemm_out<<<dim3(Bq * Nq / 64, DIMq / 64), 256, 0, stream>>>(Oh, woutt, b_out, out);
}
